// FusedLayer_18597208392431
// MI455X (gfx1250) — matrix operand order repaired and checked by reading, not compiled, not run on hardware
//
#include <hip/hip_runtime.h>

// ---------------------------------------------------------------------------
// MI455X (gfx1250) fused transformer layer, bf16 data path.
// - Weights converted fp32->bf16 once per launch; activations stored bf16.
// - GEMM: v_wmma_f32_16x16x32_bf16, 128x128x32 tiles. Tile operands are DMA'd
//   into LDS by the Tensor Data Mover (tensor_load_to_lds + s_wait_tensorcnt),
//   double-buffered; falls back to register-staged b128 copies if the TDM
//   builtin is unavailable.
// - Attention: flash-style online softmax, WMMA for QK^T and PV.
// ---------------------------------------------------------------------------

#define D_MODEL 1024
#define NHEAD   16
#define HDIM    64
#define SEQ     1024
#define BATCH   2
#define DFFN    4096
#define WINSZ   128
#define KLAT    64
#define MMEM    64

typedef __bf16 bf16_t;
typedef bf16_t vbf16 __attribute__((ext_vector_type(16)));
typedef bf16_t vbf8  __attribute__((ext_vector_type(8)));
typedef bf16_t vbf4  __attribute__((ext_vector_type(4)));
typedef float  vf8   __attribute__((ext_vector_type(8)));
typedef unsigned u32x4 __attribute__((ext_vector_type(4)));
typedef int      i32x8 __attribute__((ext_vector_type(8)));
typedef int      i32x4 __attribute__((ext_vector_type(4)));

union BF16Frag { vbf16 v; vbf8 h[2]; };
union U4H { uint4 u; bf16_t h[8]; };

static __device__ __forceinline__ bf16_t cvt_bf16(float f) { return (bf16_t)f; }

// ---------------------------------------------------------------------------
// Tensor Data Mover support detection
// ---------------------------------------------------------------------------
#if defined(__has_builtin)
#if __has_builtin(__builtin_amdgcn_tensor_load_to_lds)
#define HAVE_TDM 1
#endif
#endif
#ifndef HAVE_TDM
#define HAVE_TDM 0
#endif
#if defined(__has_include)
#if __has_include(<hip/amd_detail/amd_gfx1250_TDM.h>)
#define TDM_SIX_ARGS 1
#else
#define TDM_SIX_ARGS 0
#endif
#else
#define TDM_SIX_ARGS 0
#endif

#if HAVE_TDM
// DMA a 2D bf16 tile (128 rows x 32 elems, row stride = strideElems) from
// global memory into contiguous LDS at lds_off.  D# per CDNA5 ISA ch.8.
static __device__ __forceinline__ void tdm_load_tile(const bf16_t* gptr,
                                                     unsigned lds_off,
                                                     int strideElems)
{
  const unsigned long long ga = (unsigned long long)(const void*)gptr;
  u32x4 g0;
  g0[0] = 1u;                                            // count=1, user mode
  g0[1] = lds_off;                                       // LDS byte address
  g0[2] = (unsigned)ga;                                  // global addr lo
  g0[3] = (unsigned)((ga >> 32) & 0x1FFFFFFull) | (2u << 30);  // addr hi | type=2
  i32x8 g1;
  g1[0] = (int)(1u << 16);        // workgroup_mask=0, data_size=1 (2 bytes)
  g1[1] = (int)(32u << 16);       // tensor_dim0 = 32 (elems)
  g1[2] = (int)(128u << 16);      // tensor_dim0 hi=0 | tensor_dim1 = 128
  g1[3] = (int)(32u << 16);       // tensor_dim1 hi=0 | tile_dim0 = 32
  g1[4] = 128;                    // tile_dim1 = 128, tile_dim2 = 0
  g1[5] = strideElems;            // tensor_dim0_stride lo
  g1[6] = 0;                      // stride hi | dim1_stride lo
  g1[7] = 0;
  const i32x4 z4 = {0, 0, 0, 0};
#if TDM_SIX_ARGS
  const i32x8 z8 = {0, 0, 0, 0, 0, 0, 0, 0};
  __builtin_amdgcn_tensor_load_to_lds(g0, g1, z4, z4, z8, 0);
#else
  __builtin_amdgcn_tensor_load_to_lds(g0, g1, z4, z4, 0);
#endif
}
#endif

// ---------------------------------------------------------------------------
// fp32 -> bf16 bulk convert (weights; once per launch, vectorized)
// ---------------------------------------------------------------------------
__global__ __launch_bounds__(256)
void cvt_f32_bf16(const float* __restrict__ src, bf16_t* __restrict__ dst, int n4)
{
  const int i = blockIdx.x * 256 + threadIdx.x;
  if (i >= n4) return;
  const float4 f = *(const float4*)(src + (size_t)i * 4);
  vbf4 p;
  p[0] = cvt_bf16(f.x); p[1] = cvt_bf16(f.y);
  p[2] = cvt_bf16(f.z); p[3] = cvt_bf16(f.w);
  *(vbf4*)(dst + (size_t)i * 4) = p;
}

// ---------------------------------------------------------------------------
// RMSNorm: one block per row; src row = row % srcmod (broadcast support).
// ---------------------------------------------------------------------------
template <typename InT>
__global__ __launch_bounds__(256)
void rmsnorm_k(const InT* __restrict__ src, const float* __restrict__ w,
               bf16_t* __restrict__ dst, float* __restrict__ dst2, int srcmod)
{
  __shared__ float red[256];
  const int row = blockIdx.x;
  const int t = threadIdx.x;
  const size_t sbase = (size_t)(row % srcmod) * D_MODEL;
  const size_t dbase = (size_t)row * D_MODEL;
  float v[4]; float ss = 0.f;
#pragma unroll
  for (int i = 0; i < 4; ++i) { v[i] = (float)src[sbase + t + i * 256]; ss += v[i] * v[i]; }
  red[t] = ss; __syncthreads();
  for (int s = 128; s > 0; s >>= 1) { if (t < s) red[t] += red[t + s]; __syncthreads(); }
  const float inv = rsqrtf(red[0] * (1.0f / D_MODEL) + 1e-6f);
#pragma unroll
  for (int i = 0; i < 4; ++i) {
    const float o = w[t + i * 256] * v[i] * inv;
    dst[dbase + t + i * 256] = cvt_bf16(o);
    if (dst2) dst2[dbase + t + i * 256] = o;
  }
}

// ---------------------------------------------------------------------------
// C = alpha * (A @ W^T) + addend[row & addend_mask]   (addend rows are pow2)
// A: (Nr x Kd) bf16 row-major; W: (No x Kd) bf16 row-major (torch Linear).
// 256 thr = 8 waves; tile 128x128, K-step 32; wave -> 32x64 of C.
// ---------------------------------------------------------------------------
template <typename OutT>
__global__ __launch_bounds__(256)
void gemm_wmma_bf16(const bf16_t* __restrict__ A, const bf16_t* __restrict__ W,
                    OutT* __restrict__ C, const float* __restrict__ addend,
                    int Kd, int No, int addend_mask, float alpha)
{
  __shared__ __align__(64) bf16_t As[2][128][32];
  __shared__ __align__(64) bf16_t Bs[2][128][32];
  const int t    = threadIdx.x;
  const int lane = t & 31;
  const int wid  = t >> 5;
  const int wm   = wid & 3;        // wave row (0..3) -> 32 rows
  const int wn   = wid >> 2;       // wave col (0..1) -> 64 cols
  const int r16  = lane & 15;
  const int hi   = lane >> 4;
  const size_t rowBase = (size_t)blockIdx.y * 128;
  const size_t colBase = (size_t)blockIdx.x * 128;

  vf8 acc[2][4];
#pragma unroll
  for (int m = 0; m < 2; ++m)
#pragma unroll
    for (int n = 0; n < 4; ++n)
#pragma unroll
      for (int v = 0; v < 8; ++v) acc[m][n][v] = 0.0f;

  const int nk = Kd >> 5;

#if HAVE_TDM
  // ---- TDM double-buffered tile pipeline --------------------------------
  const unsigned ldsA0 = (unsigned)(unsigned long long)(uintptr_t)&As[0][0][0];
  const unsigned ldsA1 = (unsigned)(unsigned long long)(uintptr_t)&As[1][0][0];
  const unsigned ldsB0 = (unsigned)(unsigned long long)(uintptr_t)&Bs[0][0][0];
  const unsigned ldsB1 = (unsigned)(unsigned long long)(uintptr_t)&Bs[1][0][0];
  if (t < 32) {                       // wave 0 issues the DMAs
    tdm_load_tile(A + rowBase * (size_t)Kd, ldsA0, Kd);
    tdm_load_tile(W + colBase * (size_t)Kd, ldsB0, Kd);
    __builtin_amdgcn_s_wait_tensorcnt(0);
  }
  __syncthreads();
#else
  // ---- register-staged copy fallback ------------------------------------
  const int sr0 = t >> 2;                 // rows 0..63
  const int sr1 = sr0 + 64;               // rows 64..127
  const int sc  = (t & 3) * 8;            // 8-elem column chunk (quarter row)
  uint4 ra0, ra1, rb0, rb1;
  auto loadRegs = [&](int kk) {
    ra0 = *(const uint4*)(A + (rowBase + sr0) * (size_t)Kd + kk + sc);
    ra1 = *(const uint4*)(A + (rowBase + sr1) * (size_t)Kd + kk + sc);
    rb0 = *(const uint4*)(W + (colBase + sr0) * (size_t)Kd + kk + sc);
    rb1 = *(const uint4*)(W + (colBase + sr1) * (size_t)Kd + kk + sc);
  };
  auto storeRegs = [&](int buf) {
    *(uint4*)(&As[buf][sr0][sc]) = ra0;
    *(uint4*)(&As[buf][sr1][sc]) = ra1;
    *(uint4*)(&Bs[buf][sr0][sc]) = rb0;
    *(uint4*)(&Bs[buf][sr1][sc]) = rb1;
  };
  loadRegs(0);
  storeRegs(0);
  __syncthreads();
#endif

  for (int ki = 0; ki < nk; ++ki) {
    const int cur = ki & 1;
#if HAVE_TDM
    if (ki + 1 < nk && t < 32) {      // overlap next tile DMA with WMMAs
      const int kk = (ki + 1) << 5;
      tdm_load_tile(A + rowBase * (size_t)Kd + kk, cur ? ldsA0 : ldsA1, Kd);
      tdm_load_tile(W + colBase * (size_t)Kd + kk, cur ? ldsB0 : ldsB1, Kd);
    }
#else
    if (ki + 1 < nk) loadRegs((ki + 1) << 5);
#endif

    BF16Frag af[2];
#pragma unroll
    for (int m = 0; m < 2; ++m) {
      const int ar = wm * 32 + m * 16 + r16;
      // CDNA5 16-bit A layout: elems 0..7 -> K=hi*8+0..7, 8..15 -> K=16+hi*8+0..7
      af[m].h[0] = *(const vbf8*)(&As[cur][ar][hi * 8]);
      af[m].h[1] = *(const vbf8*)(&As[cur][ar][16 + hi * 8]);
    }
#pragma unroll
    for (int n = 0; n < 4; ++n) {
      const int br = wn * 64 + n * 16 + r16;
      BF16Frag bfr;                // B layout: lane n, K=hi*8+0..7 and 16+hi*8+0..7
      bfr.h[0] = *(const vbf8*)(&Bs[cur][br][hi * 8]);
      bfr.h[1] = *(const vbf8*)(&Bs[cur][br][16 + hi * 8]);
#pragma unroll
      for (int m = 0; m < 2; ++m)
        acc[m][n] = __builtin_amdgcn_wmma_f32_16x16x32_bf16(
            false, af[m].v, false, bfr.v, (short)0, acc[m][n], false, false);
    }

    if (ki + 1 < nk) {
#if HAVE_TDM
      if (t < 32) __builtin_amdgcn_s_wait_tensorcnt(0);
#else
      storeRegs(1 - cur);
#endif
      __syncthreads();
    }
  }

#pragma unroll
  for (int m = 0; m < 2; ++m)
#pragma unroll
    for (int n = 0; n < 4; ++n)
#pragma unroll
      for (int v = 0; v < 8; ++v) {
        const size_t row = rowBase + wm * 32 + m * 16 + hi * 8 + v;
        const size_t col = colBase + wn * 64 + n * 16 + r16;
        float out = alpha * acc[m][n][v];
        if (addend) out += addend[(size_t)((int)row & addend_mask) * (size_t)No + col];
        C[row * (size_t)No + col] = (OutT)out;
      }
}

// ---------------------------------------------------------------------------
// Flash attention, one wave per (16-query tile, head, batch); bf16 I/O.
// mode 0: dense; mode 1: |j-i|<=WIN; mode 2: |j-i|<=WIN && (j-i)%4==0.
// ---------------------------------------------------------------------------
__global__ __launch_bounds__(32)
void flash_attn_wmma(const bf16_t* __restrict__ Qp, const bf16_t* __restrict__ Kp,
                     const bf16_t* __restrict__ Vp, bf16_t* __restrict__ Op,
                     int Lq, int Lk, int mode, const unsigned char* __restrict__ kpm)
{
  __shared__ __align__(64) bf16_t Qs[16][64];
  __shared__ __align__(64) bf16_t Ks[32][64];
  __shared__ __align__(64) bf16_t VsT[64][32];   // V transposed: [hd][key]
  __shared__ __align__(64) bf16_t Ps[16][32];    // probabilities (A operand)
  const int lane = threadIdx.x;
  const int r16  = lane & 15;
  const int hi   = lane >> 4;
  const int i0   = blockIdx.x * 16;
  const int h    = blockIdx.y;
  const int b    = blockIdx.z;
  const float NEG = -3.0e38f;

  const size_t qbase = ((size_t)b * Lq + i0) * D_MODEL + h * HDIM;
#pragma unroll
  for (int i = 0; i < 4; ++i) {                  // 16 rows * 8 uint4
    const int idx = lane + i * 32;
    const int r = idx >> 3, c = (idx & 7) * 8;
    *(uint4*)(&Qs[r][c]) = *(const uint4*)(Qp + qbase + (size_t)r * D_MODEL + c);
  }

  float m_old[8], l_sum[8];
  vf8 accO[4];
#pragma unroll
  for (int v = 0; v < 8; ++v) { m_old[v] = NEG; l_sum[v] = 0.f; }
#pragma unroll
  for (int n = 0; n < 4; ++n)
#pragma unroll
    for (int v = 0; v < 8; ++v) accO[n][v] = 0.f;

  int kb0 = 0, kb1 = Lk;
  if (mode != 0) {
    kb0 = i0 - WINSZ; if (kb0 < 0) kb0 = 0; kb0 &= ~31;
    kb1 = i0 + 15 + WINSZ + 1; if (kb1 > Lk) kb1 = Lk; kb1 = (kb1 + 31) & ~31;
  }

  for (int kb = kb0; kb < kb1; kb += 32) {
    const size_t kvbase = ((size_t)b * Lk + kb) * D_MODEL + h * HDIM;
    // stage all 16 b128 loads before any LDS store (pipelined loads)
    uint4 kreg[8]; U4H vreg[8];
#pragma unroll
    for (int i = 0; i < 8; ++i) {                // 32 rows * 8 uint4
      const int idx = lane + i * 32;
      const int r = idx >> 3, c = (idx & 7) * 8;
      kreg[i]   = *(const uint4*)(Kp + kvbase + (size_t)r * D_MODEL + c);
      vreg[i].u = *(const uint4*)(Vp + kvbase + (size_t)r * D_MODEL + c);
    }
#pragma unroll
    for (int i = 0; i < 8; ++i) {
      const int idx = lane + i * 32;
      const int r = idx >> 3, c = (idx & 7) * 8;
      *(uint4*)(&Ks[r][c]) = kreg[i];
#pragma unroll
      for (int j = 0; j < 8; ++j) VsT[c + j][r] = vreg[i].h[j];
    }
    __syncthreads();

    // S = (Q K^T) / 8 : two 16x16 C fragments over 32 keys, K-dim 64 (2 steps)
    vf8 accS[2];
#pragma unroll
    for (int n = 0; n < 2; ++n)
#pragma unroll
      for (int v = 0; v < 8; ++v) accS[n][v] = 0.f;
#pragma unroll
    for (int kc = 0; kc < 2; ++kc) {
      BF16Frag a;
      a.h[0] = *(const vbf8*)(&Qs[r16][kc * 32 + hi * 8]);
      a.h[1] = *(const vbf8*)(&Qs[r16][kc * 32 + 16 + hi * 8]);
#pragma unroll
      for (int n = 0; n < 2; ++n) {
        BF16Frag bb;
        bb.h[0] = *(const vbf8*)(&Ks[n * 16 + r16][kc * 32 + hi * 8]);
        bb.h[1] = *(const vbf8*)(&Ks[n * 16 + r16][kc * 32 + 16 + hi * 8]);
        accS[n] = __builtin_amdgcn_wmma_f32_16x16x32_bf16(
            false, a.v, false, bb.v, (short)0, accS[n], false, false);
      }
    }

    // mask + online softmax (row = hi*8+v, spread over a 16-lane group)
#pragma unroll
    for (int v = 0; v < 8; ++v) {
      const int qi  = i0 + hi * 8 + v;
      const int kj0 = kb + r16;
      const int kj1 = kb + 16 + r16;
      float s0 = accS[0][v] * 0.125f;
      float s1 = accS[1][v] * 0.125f;
      const int d0 = kj0 - qi, d1 = kj1 - qi;
      if (mode == 1) {
        if (d0 < -WINSZ || d0 > WINSZ) s0 = NEG;
        if (d1 < -WINSZ || d1 > WINSZ) s1 = NEG;
      } else if (mode == 2) {
        if (d0 < -WINSZ || d0 > WINSZ || (d0 & 3)) s0 = NEG;
        if (d1 < -WINSZ || d1 > WINSZ || (d1 & 3)) s1 = NEG;
      }
      if (kpm) {
        if (kpm[(size_t)b * Lk + kj0]) s0 = NEG;
        if (kpm[(size_t)b * Lk + kj1]) s1 = NEG;
      }
      float mx = fmaxf(s0, s1);
#pragma unroll
      for (int off = 1; off < 16; off <<= 1) mx = fmaxf(mx, __shfl_xor(mx, off, 32));
      const float m_new = fmaxf(m_old[v], mx);
      const float p0 = __expf(s0 - m_new);
      const float p1 = __expf(s1 - m_new);
      float ps = p0 + p1;
#pragma unroll
      for (int off = 1; off < 16; off <<= 1) ps += __shfl_xor(ps, off, 32);
      const float corr = __expf(m_old[v] - m_new);
      l_sum[v] = l_sum[v] * corr + ps;
      m_old[v] = m_new;
#pragma unroll
      for (int n = 0; n < 4; ++n) accO[n][v] *= corr;
      Ps[hi * 8 + v][r16]      = cvt_bf16(p0);
      Ps[hi * 8 + v][16 + r16] = cvt_bf16(p1);
    }
    __syncthreads();

    // O += P @ V : P as A operand (16x32 keys), V^T rows -> contiguous B frags
    BF16Frag pa;
    pa.h[0] = *(const vbf8*)(&Ps[r16][hi * 8]);
    pa.h[1] = *(const vbf8*)(&Ps[r16][16 + hi * 8]);
#pragma unroll
    for (int n = 0; n < 4; ++n) {
      BF16Frag bb;
      bb.h[0] = *(const vbf8*)(&VsT[n * 16 + r16][hi * 8]);
      bb.h[1] = *(const vbf8*)(&VsT[n * 16 + r16][16 + hi * 8]);
      accO[n] = __builtin_amdgcn_wmma_f32_16x16x32_bf16(
          false, pa.v, false, bb.v, (short)0, accO[n], false, false);
    }
    __syncthreads();
  }

  const size_t obase = ((size_t)b * Lq + i0) * D_MODEL + h * HDIM;
#pragma unroll
  for (int n = 0; n < 4; ++n)
#pragma unroll
    for (int v = 0; v < 8; ++v)
      Op[obase + (size_t)(hi * 8 + v) * D_MODEL + n * 16 + r16] =
          cvt_bf16(accO[n][v] / (l_sum[v] + 1e-30f));
}

// ---------------------------------------------------------------------------
// Gate softmax over 4 branches + fused residual:  xnew = x + sum_g w_g * H_g
// ---------------------------------------------------------------------------
__global__ __launch_bounds__(256)
void gate_fuse(const float* __restrict__ x, const float* __restrict__ hf,
               const float* __restrict__ gw, const float* __restrict__ gb,
               const bf16_t* __restrict__ b0, const bf16_t* __restrict__ b1,
               const bf16_t* __restrict__ b2, const bf16_t* __restrict__ b3,
               float* __restrict__ xnew)
{
  __shared__ float red[256];
  __shared__ float wsm[4];
  const int row = blockIdx.x, t = threadIdx.x;
  const size_t base = (size_t)row * D_MODEL;
  float hv[4];
#pragma unroll
  for (int i = 0; i < 4; ++i) hv[i] = hf[base + t + i * 256];
  for (int g = 0; g < 4; ++g) {
    float p = 0.f;
#pragma unroll
    for (int i = 0; i < 4; ++i) p += hv[i] * gw[(size_t)g * D_MODEL + t + i * 256];
    red[t] = p; __syncthreads();
    for (int s = 128; s > 0; s >>= 1) { if (t < s) red[t] += red[t + s]; __syncthreads(); }
    if (t == 0) wsm[g] = red[0] + gb[g];
    __syncthreads();
  }
  if (t == 0) {
    float mx = fmaxf(fmaxf(wsm[0], wsm[1]), fmaxf(wsm[2], wsm[3]));
    float e0 = __expf(wsm[0] - mx), e1 = __expf(wsm[1] - mx);
    float e2 = __expf(wsm[2] - mx), e3 = __expf(wsm[3] - mx);
    const float inv = 1.0f / (e0 + e1 + e2 + e3);
    wsm[0] = e0 * inv; wsm[1] = e1 * inv; wsm[2] = e2 * inv; wsm[3] = e3 * inv;
  }
  __syncthreads();
#pragma unroll
  for (int i = 0; i < 4; ++i) {
    const size_t c = base + t + i * 256;
    xnew[c] = x[c] + wsm[0] * (float)b0[c] + wsm[1] * (float)b1[c] +
                     wsm[2] * (float)b2[c] + wsm[3] * (float)b3[c];
  }
}

// ---------------------------------------------------------------------------
// SwiGLU activation: t[row,c] = silu(u[row,c]) * u[row, DFFN + c]
// ---------------------------------------------------------------------------
__global__ __launch_bounds__(256)
void silu_mul(const bf16_t* __restrict__ u, bf16_t* __restrict__ t, int n)
{
  const int i = blockIdx.x * 256 + threadIdx.x;
  if (i >= n) return;
  const int row = i >> 12;         // / DFFN (4096)
  const int c   = i & (DFFN - 1);
  const float a = (float)u[(size_t)row * (2 * DFFN) + c];
  const float g = (float)u[(size_t)row * (2 * DFFN) + DFFN + c];
  t[i] = cvt_bf16((a / (1.0f + __expf(-a))) * g);
}

// ---------------------------------------------------------------------------
extern "C" void kernel_launch(void* const* d_in, const int* in_sizes, int n_in,
                              void* d_out, int out_size, void* d_ws, size_t ws_size,
                              hipStream_t stream)
{
  (void)in_sizes; (void)n_in; (void)out_size; (void)ws_size;
  const float* x          = (const float*)d_in[0];
  const unsigned char* kp = (const unsigned char*)d_in[1];
  const float* norm_w     = (const float*)d_in[2];
  const float* ld_nq      = (const float*)d_in[3];
  const float* ld_nk      = (const float*)d_in[4];
  const float* local_w    = (const float*)d_in[5];
  const float* dil_w      = (const float*)d_in[6];
  const float* lat_tok    = (const float*)d_in[7];
  const float* lp_nq      = (const float*)d_in[8];
  const float* lp_nk      = (const float*)d_in[9];
  const float* lat_to_w   = (const float*)d_in[10];
  const float* lat_from_w = (const float*)d_in[11];
  const float* mem_tok    = (const float*)d_in[12];
  const float* mem_nq     = (const float*)d_in[13];
  const float* mem_nk     = (const float*)d_in[14];
  const float* mem_read_w = (const float*)d_in[15];
  const float* mem_write_w= (const float*)d_in[16];
  const float* gate_w     = (const float*)d_in[17];
  const float* gate_b     = (const float*)d_in[18];
  const float* up_w       = (const float*)d_in[19];
  const float* down_w     = (const float*)d_in[20];
  float* out = (float*)d_out;

  const size_t DD = (size_t)D_MODEL * D_MODEL;       // 1,048,576
  const size_t S  = (size_t)BATCH * SEQ * D_MODEL;   // 2,097,152
  const size_t SM = (size_t)BATCH * KLAT * D_MODEL;  // 131,072

  // ---- workspace carve (bytes), fixed & deterministic --------------------
  char* p = (char*)d_ws;
  auto alloc = [&](size_t bytes) -> char* {
    char* r = p; p += (bytes + 255) & ~(size_t)255; return r;
  };
  bf16_t* wb_loc  = (bf16_t*)alloc(4 * DD * 2);
  bf16_t* wb_dil  = (bf16_t*)alloc(4 * DD * 2);
  bf16_t* wb_lto  = (bf16_t*)alloc(4 * DD * 2);
  bf16_t* wb_lfr  = (bf16_t*)alloc(4 * DD * 2);
  bf16_t* wb_mrd  = (bf16_t*)alloc(4 * DD * 2);
  bf16_t* wb_mwr  = (bf16_t*)alloc(4 * DD * 2);
  bf16_t* wb_up   = (bf16_t*)alloc(8 * DD * 2);
  bf16_t* wb_down = (bf16_t*)alloc(4 * DD * 2);
  bf16_t* h_bf    = (bf16_t*)alloc(S * 2);
  float*  h_f32   = (float*) alloc(S * 4);
  bf16_t* qn      = (bf16_t*)alloc(S * 2);
  bf16_t* kn      = (bf16_t*)alloc(S * 2);
  bf16_t* Qb      = (bf16_t*)alloc(S * 2);
  bf16_t* Kb      = (bf16_t*)alloc(S * 2);
  bf16_t* Vb      = (bf16_t*)alloc(S * 2);
  bf16_t* attnb   = (bf16_t*)alloc(S * 2);
  bf16_t* o_loc   = (bf16_t*)alloc(S * 2);
  bf16_t* o_dil   = (bf16_t*)alloc(S * 2);
  bf16_t* x_lat   = (bf16_t*)alloc(S * 2);
  bf16_t* x_mem   = (bf16_t*)alloc(S * 2);
  float*  x_new   = (float*) alloc(S * 4);
  bf16_t* ubuf    = (bf16_t*)alloc((size_t)BATCH * SEQ * 2 * DFFN * 2);
  bf16_t* tbuf    = (bf16_t*)alloc((size_t)BATCH * SEQ * DFFN * 2);
  bf16_t* s0      = (bf16_t*)alloc(SM * 2);
  bf16_t* s1      = (bf16_t*)alloc(SM * 2);
  bf16_t* s2      = (bf16_t*)alloc(SM * 2);
  bf16_t* s3      = (bf16_t*)alloc(SM * 2);
  bf16_t* s4      = (bf16_t*)alloc(SM * 2);
  float*  lat     = out + S;        // (B,K,D) fp32, final output slot
  float*  mem     = out + S + SM;   // (B,M,D) fp32, final output slot

  const int NR = BATCH * SEQ;    // 2048
  const int NS = BATCH * KLAT;   // 128

  auto conv = [&](const float* src, bf16_t* dst, size_t n) {
    const int n4 = (int)(n / 4);
    cvt_f32_bf16<<<dim3((n4 + 255) / 256), dim3(256), 0, stream>>>(src, dst, n4);
  };
  auto gemmB = [&](const bf16_t* A, const bf16_t* W, bf16_t* C, int Nr, int Kd, int No,
                   const float* add, int addrows, float alpha) {
    gemm_wmma_bf16<bf16_t><<<dim3(No / 128, Nr / 128), dim3(256), 0, stream>>>(
        A, W, C, add, Kd, No, addrows - 1, alpha);
  };
  auto gemmF = [&](const bf16_t* A, const bf16_t* W, float* C, int Nr, int Kd, int No,
                   const float* add, int addrows, float alpha) {
    gemm_wmma_bf16<float><<<dim3(No / 128, Nr / 128), dim3(256), 0, stream>>>(
        A, W, C, add, Kd, No, addrows - 1, alpha);
  };
  auto attn = [&](const bf16_t* Q, const bf16_t* Kk, const bf16_t* Vv, bf16_t* O,
                  int Lq, int Lk, int mode, const unsigned char* kpm) {
    flash_attn_wmma<<<dim3(Lq / 16, NHEAD, BATCH), dim3(32), 0, stream>>>(
        Q, Kk, Vv, O, Lq, Lk, mode, kpm);
  };

  // ---- weight conversion (once per launch) ------------------------------
  conv(local_w,     wb_loc,  4 * DD);
  conv(dil_w,       wb_dil,  4 * DD);
  conv(lat_to_w,    wb_lto,  4 * DD);
  conv(lat_from_w,  wb_lfr,  4 * DD);
  conv(mem_read_w,  wb_mrd,  4 * DD);
  conv(mem_write_w, wb_mwr,  4 * DD);
  conv(up_w,        wb_up,   8 * DD);
  conv(down_w,      wb_down, 4 * DD);

  // ---- pre-norms --------------------------------------------------------
  rmsnorm_k<float><<<dim3(NR), dim3(256), 0, stream>>>(x, norm_w, h_bf, h_f32, NR);
  rmsnorm_k<bf16_t><<<dim3(NR), dim3(256), 0, stream>>>(h_bf, ld_nq, qn, nullptr, NR);
  rmsnorm_k<bf16_t><<<dim3(NR), dim3(256), 0, stream>>>(h_bf, ld_nk, kn, nullptr, NR);

  // ---- local branch -----------------------------------------------------
  gemmB(qn, wb_loc + 0 * DD, Qb, NR, D_MODEL, D_MODEL, nullptr, 1, 1.0f);
  gemmB(kn, wb_loc + 1 * DD, Kb, NR, D_MODEL, D_MODEL, nullptr, 1, 1.0f);
  gemmB(kn, wb_loc + 2 * DD, Vb, NR, D_MODEL, D_MODEL, nullptr, 1, 1.0f);
  attn(Qb, Kb, Vb, attnb, SEQ, SEQ, 1, kp);
  gemmB(attnb, wb_loc + 3 * DD, o_loc, NR, D_MODEL, D_MODEL, nullptr, 1, 1.0f);

  // ---- dilated branch ---------------------------------------------------
  gemmB(qn, wb_dil + 0 * DD, Qb, NR, D_MODEL, D_MODEL, nullptr, 1, 1.0f);
  gemmB(kn, wb_dil + 1 * DD, Kb, NR, D_MODEL, D_MODEL, nullptr, 1, 1.0f);
  gemmB(kn, wb_dil + 2 * DD, Vb, NR, D_MODEL, D_MODEL, nullptr, 1, 1.0f);
  attn(Qb, Kb, Vb, attnb, SEQ, SEQ, 2, kp);
  gemmB(attnb, wb_dil + 3 * DD, o_dil, NR, D_MODEL, D_MODEL, nullptr, 1, 1.0f);

  // ---- latent pool: latents read tokens ---------------------------------
  rmsnorm_k<bf16_t><<<dim3(NR), dim3(256), 0, stream>>>(h_bf, lp_nk, kn, nullptr, NR);
  rmsnorm_k<float><<<dim3(NS), dim3(256), 0, stream>>>(lat_tok, lp_nq, s0, nullptr, KLAT);
  gemmB(s0, wb_lto + 0 * DD, s1, NS, D_MODEL, D_MODEL, nullptr, 1, 1.0f);
  gemmB(kn, wb_lto + 1 * DD, Kb, NR, D_MODEL, D_MODEL, nullptr, 1, 1.0f);
  gemmB(kn, wb_lto + 2 * DD, Vb, NR, D_MODEL, D_MODEL, nullptr, 1, 1.0f);
  attn(s1, Kb, Vb, s4, KLAT, SEQ, 0, kp);
  gemmF(s4, wb_lto + 3 * DD, lat, NS, D_MODEL, D_MODEL, lat_tok, KLAT, 1.0f);

  // ---- latent pool: tokens read latents ---------------------------------
  rmsnorm_k<bf16_t><<<dim3(NR), dim3(256), 0, stream>>>(h_bf, lp_nq, qn, nullptr, NR);
  rmsnorm_k<float><<<dim3(NS), dim3(256), 0, stream>>>(lat, lp_nk, s0, nullptr, NS);
  gemmB(qn, wb_lfr + 0 * DD, Qb, NR, D_MODEL, D_MODEL, nullptr, 1, 1.0f);
  gemmB(s0, wb_lfr + 1 * DD, s2, NS, D_MODEL, D_MODEL, nullptr, 1, 1.0f);
  gemmB(s0, wb_lfr + 2 * DD, s3, NS, D_MODEL, D_MODEL, nullptr, 1, 1.0f);
  attn(Qb, s2, s3, attnb, SEQ, KLAT, 0, nullptr);
  gemmB(attnb, wb_lfr + 3 * DD, x_lat, NR, D_MODEL, D_MODEL, h_f32, NR, 1.0f);

  // ---- infini-memory: write (EMA fused: mem = mem_tok + 0.1 * mha) -------
  rmsnorm_k<bf16_t><<<dim3(NR), dim3(256), 0, stream>>>(h_bf, mem_nk, kn, nullptr, NR);
  rmsnorm_k<float><<<dim3(NS), dim3(256), 0, stream>>>(mem_tok, mem_nq, s0, nullptr, MMEM);
  gemmB(s0, wb_mwr + 0 * DD, s1, NS, D_MODEL, D_MODEL, nullptr, 1, 1.0f);
  gemmB(kn, wb_mwr + 1 * DD, Kb, NR, D_MODEL, D_MODEL, nullptr, 1, 1.0f);
  gemmB(kn, wb_mwr + 2 * DD, Vb, NR, D_MODEL, D_MODEL, nullptr, 1, 1.0f);
  attn(s1, Kb, Vb, s4, MMEM, SEQ, 0, kp);
  gemmF(s4, wb_mwr + 3 * DD, mem, NS, D_MODEL, D_MODEL, mem_tok, MMEM, 0.1f);

  // ---- infini-memory: read ----------------------------------------------
  rmsnorm_k<bf16_t><<<dim3(NR), dim3(256), 0, stream>>>(h_bf, mem_nq, qn, nullptr, NR);
  rmsnorm_k<float><<<dim3(NS), dim3(256), 0, stream>>>(mem, mem_nk, s0, nullptr, NS);
  gemmB(qn, wb_mrd + 0 * DD, Qb, NR, D_MODEL, D_MODEL, nullptr, 1, 1.0f);
  gemmB(s0, wb_mrd + 1 * DD, s2, NS, D_MODEL, D_MODEL, nullptr, 1, 1.0f);
  gemmB(s0, wb_mrd + 2 * DD, s3, NS, D_MODEL, D_MODEL, nullptr, 1, 1.0f);
  attn(Qb, s2, s3, attnb, SEQ, KLAT, 0, nullptr);
  gemmB(attnb, wb_mrd + 3 * DD, x_mem, NR, D_MODEL, D_MODEL, h_f32, NR, 1.0f);

  // ---- gated fusion -----------------------------------------------------
  gate_fuse<<<dim3(NR), dim3(256), 0, stream>>>(x, h_f32, gate_w, gate_b,
                                                o_loc, o_dil, x_lat, x_mem, x_new);

  // ---- SwiGLU FFN -------------------------------------------------------
  rmsnorm_k<float><<<dim3(NR), dim3(256), 0, stream>>>(x_new, norm_w, h_bf, nullptr, NR);
  gemmB(h_bf, wb_up, ubuf, NR, D_MODEL, 2 * DFFN, nullptr, 1, 1.0f);
  {
    const int n = NR * DFFN;
    silu_mul<<<dim3((n + 255) / 256), dim3(256), 0, stream>>>(ubuf, tbuf, n);
  }
  gemmF(tbuf, wb_down, out, NR, DFFN, D_MODEL, x_new, NR, 1.0f);
}
